// Head_52750788329975
// MI455X (gfx1250) — matrix operand order repaired and checked by reading, not compiled, not run on hardware
//
#include <hip/hip_runtime.h>
#include <hip/hip_bf16.h>
#include <math.h>

#define B_ 16
#define T_ 2048
#define C_ 128
#define H_ 128
#define WAVES_ 4            // waves per attention block
#define KROWS_ 136          // K LDS row stride in elems (128 + 8 pad) -> 272B
#define VROWS_ 72           // V LDS row stride in elems (64 + 8 pad)  -> 144B

typedef __attribute__((ext_vector_type(16))) __bf16 v16bf;
typedef __attribute__((ext_vector_type(8)))  float  v8f;
typedef __attribute__((ext_vector_type(16))) unsigned short us16;
typedef __attribute__((ext_vector_type(8)))  unsigned short us8;
typedef __attribute__((ext_vector_type(4)))  float  f4;
typedef __attribute__((ext_vector_type(4)))  unsigned int u32x4;
typedef __attribute__((ext_vector_type(8)))  int i32x8;
typedef __attribute__((ext_vector_type(4)))  int i32x4;

union U16 { us16 v; us8 h[2]; };

#if defined(__has_builtin)
#if __has_builtin(__builtin_amdgcn_tensor_load_to_lds) && \
    __has_builtin(__builtin_amdgcn_s_wait_tensorcnt)
#define HAVE_TDM 1
#endif
#endif

static __device__ __forceinline__ unsigned short f2bf(float f) {
  unsigned int u = __builtin_bit_cast(unsigned int, f);
  u += 0x7fffu + ((u >> 16) & 1u);          // round-to-nearest-even
  return (unsigned short)(u >> 16);
}

static __device__ __forceinline__ v8f wmma_bf16(us16 a, us16 b, v8f c) {
  return __builtin_amdgcn_wmma_f32_16x16x32_bf16(
      false, __builtin_bit_cast(v16bf, a),
      false, __builtin_bit_cast(v16bf, b),
      (short)0, c, false, false);
}

// LDS byte offset of a __shared__ object: generic LDS addresses carry the
// workgroup-relative offset in the low 32 bits (aperture layout).
static __device__ __forceinline__ unsigned lds_off(const void* p) {
  return (unsigned)(unsigned long long)(size_t)p;
}

#ifdef HAVE_TDM
// 2D TDM tile load: tensor of dim0 x dim1 (elems, 2B each), row stride
// stride0 (elems), tile of tile0 x tile1 starting at gaddr, into LDS with
// row padding (pad_interval code padi, pad_amount code pada -> +16B/row).
static __device__ __forceinline__ void tdm_load_2d(
    unsigned lds, const void* gaddr,
    unsigned dim0, unsigned dim1, unsigned tile0, unsigned tile1,
    unsigned long long stride0, unsigned padi, unsigned pada) {
  unsigned long long ga = (unsigned long long)(size_t)gaddr;
  u32x4 g0;
  g0[0] = 1u;                                          // count=1
  g0[1] = lds;                                         // lds_addr
  g0[2] = (unsigned)(ga & 0xffffffffu);                // global_addr[31:0]
  g0[3] = (unsigned)((ga >> 32) & 0x1ffffffu) | (2u << 30);  // ga[56:32], type=2
  i32x8 g1;
  g1[0] = (int)((1u << 16) |                            // data_size = 2B
                (1u << 20) |                            // pad_enable
                (padi << 22) | (pada << 25));
  g1[1] = (int)((dim0 & 0xffffu) << 16);                // tensor_dim0[15:0]
  g1[2] = (int)(((dim0 >> 16) & 0xffffu) | ((dim1 & 0xffffu) << 16));
  g1[3] = (int)(((dim1 >> 16) & 0xffffu) | ((tile0 & 0xffffu) << 16));
  g1[4] = (int)(tile1 & 0xffffu);                       // tile_dim1, tile_dim2=0
  g1[5] = (int)(unsigned)(stride0 & 0xffffffffu);       // dim0_stride[31:0]
  g1[6] = (int)(unsigned)((stride0 >> 32) & 0xffffu);   // dim0_stride[47:32]
  g1[7] = 0;
  i32x4 z4 = (i32x4)0;
#if __clang_major__ >= 23
  i32x8 z8 = (i32x8)0;
  __builtin_amdgcn_tensor_load_to_lds(g0, g1, z4, z4, z8, 0);
#else
  __builtin_amdgcn_tensor_load_to_lds(g0, g1, z4, z4, 0);
#endif
}
#endif

// ---------------- prep kernels ----------------

__global__ void convert_x_kernel(const float* __restrict__ x,
                                 unsigned short* __restrict__ xb, int n) {
  int i = blockIdx.x * blockDim.x + threadIdx.x;
  if (i < n) xb[i] = f2bf(x[i]);
}

// Wt[h][c] = W[c][h]; Wq gets the extra C^-0.5 attention scale folded in.
__global__ void prep_w_kernel(const float* __restrict__ Wq,
                              const float* __restrict__ Wk,
                              const float* __restrict__ Wv,
                              unsigned short* __restrict__ Wqt,
                              unsigned short* __restrict__ Wkt,
                              unsigned short* __restrict__ Wvt) {
  int i = blockIdx.x * blockDim.x + threadIdx.x;   // i = h*C + c
  if (i >= H_ * C_) return;
  int h = i / C_, c = i % C_;
  const float scale = 0.08838834764831845f;        // 1/sqrt(128)
  Wqt[i] = f2bf(Wq[c * H_ + h] * scale);
  Wkt[i] = f2bf(Wk[c * H_ + h]);
  Wvt[i] = f2bf(Wv[c * H_ + h]);
}

// ---------------- QKV projection (WMMA) ----------------

__global__ __launch_bounds__(128) void proj_kernel(
    const unsigned short* __restrict__ xb,
    const unsigned short* __restrict__ Wqt,
    const unsigned short* __restrict__ Wkt,
    const unsigned short* __restrict__ Wvt,
    unsigned short* __restrict__ Qb,
    unsigned short* __restrict__ Kb,
    unsigned short* __restrict__ Vt) {
  int lane = threadIdx.x & 31;
  int wave = threadIdx.x >> 5;
  int tile = blockIdx.x * 4 + wave;      // 16-token tile
  int r0 = tile * 16;
  int m_ = lane & 15;
  int half = lane >> 4;

  us16 ax[4];
  {
    const unsigned short* xp = xb + (size_t)(r0 + m_) * C_ + half * 16;
#pragma unroll
    for (int kk = 0; kk < 4; ++kk) {
      U16 u;
      u.h[0] = *(const us8*)(xp + kk * 32);
      u.h[1] = *(const us8*)(xp + kk * 32 + 8);
      ax[kk] = u.v;
    }
  }

  const unsigned short* Ws[3] = {Wqt, Wkt, Wvt};
#pragma unroll
  for (int mat = 0; mat < 3; ++mat) {
    const unsigned short* W = Ws[mat];
#pragma unroll
    for (int ht = 0; ht < 8; ++ht) {
      int h0 = ht * 16;
      v8f acc = {};
#pragma unroll
      for (int kk = 0; kk < 4; ++kk) {
        const unsigned short* wp =
            W + (size_t)(h0 + m_) * C_ + kk * 32 + half * 16;
        us16 bw = *(const us16*)wp;
        acc = wmma_bf16(ax[kk], bw, acc);
      }
      int h = h0 + m_;
#pragma unroll
      for (int v = 0; v < 8; ++v) {
        int token = r0 + v + 8 * half;
        unsigned short val = f2bf(acc[v]);
        if (mat == 0)      Qb[(size_t)token * H_ + h] = val;
        else if (mat == 1) Kb[(size_t)token * H_ + h] = val;
        else {
          int bidx = token >> 11;
          int tl = token & (T_ - 1);
          Vt[(size_t)bidx * H_ * T_ + (size_t)h * T_ + tl] = val;
        }
      }
    }
  }
}

// ---------------- flash attention ----------------
// 4 waves/block, 32 queries/wave (two 16-query tiles sharing K/V fragments).
// 64-key K and V^T tiles double-buffered in LDS via the Tensor Data Mover
// (padded rows -> bank-conflict-free ds_load_b128 fragment reads).

__global__ __launch_bounds__(128) void attn_kernel(
    const unsigned short* __restrict__ Qb,
    const unsigned short* __restrict__ Kb,
    const unsigned short* __restrict__ Vt,
    const unsigned char* __restrict__ pmask,
    float* __restrict__ out) {
  __shared__ alignas(16) unsigned short sK[2][64 * KROWS_];
  __shared__ alignas(16) unsigned short sV[2][128 * VROWS_];

  int lane = threadIdx.x & 31;
  int wave = threadIdx.x >> 5;
  int b = blockIdx.y;
  int q0 = (blockIdx.x * WAVES_ + wave) * 32;
  int m_ = lane & 15;
  int half = lane >> 4;

  // Q as B-fragments for both query tiles (lane n = query, 16 contiguous h)
  us16 qf[2][4];
#pragma unroll
  for (int j = 0; j < 2; ++j) {
    const unsigned short* qp =
        Qb + ((size_t)(b * T_) + q0 + j * 16 + m_) * H_ + half * 16;
#pragma unroll
    for (int kk = 0; kk < 4; ++kk) qf[j][kk] = *(const us16*)(qp + kk * 32);
  }

  v8f acc[2][8];
#pragma unroll
  for (int j = 0; j < 2; ++j)
#pragma unroll
    for (int ht = 0; ht < 8; ++ht) acc[j][ht] = (v8f){};
  float mrow[2] = {-INFINITY, -INFINITY};
  float lrow[2] = {0.0f, 0.0f};

  const int NIT = T_ / 64;

#ifdef HAVE_TDM
  // prologue: stage tile 0 into buffer 0 (one wave issues for the block)
  if (wave == 0) {
    tdm_load_2d(lds_off(&sK[0][0]), Kb + ((size_t)(b * T_)) * H_,
                128, 64, 128, 64, 128ull, 5, 3);
    tdm_load_2d(lds_off(&sV[0][0]), Vt + (size_t)b * H_ * T_,
                64, 128, 64, 128, (unsigned long long)T_, 4, 3);
  }
#endif

  for (int it = 0; it < NIT; ++it) {
    int buf = it & 1;
    int s0 = it * 64;
    __syncthreads();   // everyone done reading buf^1 (tile it-1)
#ifdef HAVE_TDM
    if (wave == 0) {
      if (it + 1 < NIT) {
        int s1 = s0 + 64;
        tdm_load_2d(lds_off(&sK[buf ^ 1][0]),
                    Kb + ((size_t)(b * T_) + s1) * H_,
                    128, 64, 128, 64, 128ull, 5, 3);
        tdm_load_2d(lds_off(&sV[buf ^ 1][0]),
                    Vt + (size_t)b * H_ * T_ + s1,
                    64, 128, 64, 128, (unsigned long long)T_, 4, 3);
        __builtin_amdgcn_s_wait_tensorcnt((short)2);   // tile it arrived
      } else {
        __builtin_amdgcn_s_wait_tensorcnt((short)0);
      }
    }
#else
    // synchronous cooperative staging fallback (128 threads)
    {
      int tid = threadIdx.x;
      const unsigned short* kbase = Kb + ((size_t)(b * T_) + s0) * H_;
#pragma unroll
      for (int c = 0; c < 8; ++c) {
        int ch = tid + c * 128;            // 1024 chunks of 8 elems (K)
        int r = ch >> 4, cc = ch & 15;
        *(us8*)&sK[buf][r * KROWS_ + cc * 8] =
            *(const us8*)(kbase + (size_t)r * H_ + cc * 8);
      }
      const unsigned short* vbase = Vt + (size_t)b * H_ * T_ + s0;
#pragma unroll
      for (int c = 0; c < 8; ++c) {
        int ch = tid + c * 128;            // 1024 chunks of 8 elems (V)
        int r = ch >> 3, cc = ch & 7;
        *(us8*)&sV[buf][r * VROWS_ + cc * 8] =
            *(const us8*)(vbase + (size_t)r * T_ + cc * 8);
      }
    }
#endif
    __syncthreads();   // tile it visible to all waves

#pragma unroll
    for (int sub = 0; sub < 2; ++sub) {
      int ss = sub * 32;
      unsigned char mb = pmask[b * T_ + s0 + ss + lane];
      unsigned int vmask = (unsigned int)__ballot(mb != 0);

      // S^T tiles for both query tiles; K fragment loaded once, used twice
      v8f st[2][2];
      st[0][0] = (v8f){}; st[0][1] = (v8f){};
      st[1][0] = (v8f){}; st[1][1] = (v8f){};
#pragma unroll
      for (int t = 0; t < 2; ++t) {
#pragma unroll
        for (int kk = 0; kk < 4; ++kk) {
          const unsigned short* kp =
              &sK[buf][(ss + t * 16 + m_) * KROWS_ + kk * 32 + half * 16];
          U16 u;
          u.h[0] = *(const us8*)kp;
          u.h[1] = *(const us8*)(kp + 8);
          st[0][t] = wmma_bf16(u.v, qf[0][kk], st[0][t]);
          st[1][t] = wmma_bf16(u.v, qf[1][kk], st[1][t]);
        }
      }

      us16 pf[2];
#pragma unroll
      for (int j = 0; j < 2; ++j) {
        float rmax = -INFINITY;
#pragma unroll
        for (int t = 0; t < 2; ++t)
#pragma unroll
          for (int v = 0; v < 8; ++v) {
            int key = t * 16 + v + 8 * half;
            float x = ((vmask >> key) & 1u) ? st[j][t][v] : -INFINITY;
            st[j][t][v] = x;
            rmax = fmaxf(rmax, x);
          }
        rmax = fmaxf(rmax, __shfl_xor(rmax, 16));
        float mnew = fmaxf(mrow[j], rmax);
        float alpha = __expf(mrow[j] - mnew);
        float p0[8], p1[8];
        float ps = 0.0f;
#pragma unroll
        for (int v = 0; v < 8; ++v) {
          p0[v] = __expf(st[j][0][v] - mnew);
          p1[v] = __expf(st[j][1][v] - mnew);
          ps += p0[v] + p1[v];
        }
        ps += __shfl_xor(ps, 16);
        lrow[j] = lrow[j] * alpha + ps;
        mrow[j] = mnew;
#pragma unroll
        for (int ht = 0; ht < 8; ++ht)
#pragma unroll
          for (int v = 0; v < 8; ++v) acc[j][ht][v] *= alpha;
        // P^T B-fragment via lane-16 exchange
#pragma unroll
        for (int v = 0; v < 8; ++v) {
          float x0 = __shfl_xor(p0[v], 16);
          float x1 = __shfl_xor(p1[v], 16);
          pf[j][v]     = f2bf(half ? x1 : p0[v]);
          pf[j][v + 8] = f2bf(half ? p1[v] : x0);
        }
      }

      // O^T += V^T x P^T ; V fragment loaded once, used for both query tiles
#pragma unroll
      for (int ht = 0; ht < 8; ++ht) {
        const unsigned short* vp =
            &sV[buf][(ht * 16 + m_) * VROWS_ + ss + half * 16];
        U16 u;
        u.h[0] = *(const us8*)vp;
        u.h[1] = *(const us8*)(vp + 8);
        acc[0][ht] = wmma_bf16(u.v, pf[0], acc[0][ht]);
        acc[1][ht] = wmma_bf16(u.v, pf[1], acc[1][ht]);
      }
    }
  }

  // normalize + store
#pragma unroll
  for (int j = 0; j < 2; ++j) {
    float rl = 1.0f / lrow[j];
    size_t row = ((size_t)(b * T_) + q0 + j * 16 + m_) * H_;
#pragma unroll
    for (int ht = 0; ht < 8; ++ht) {
      size_t idx = row + ht * 16 + half * 8;
      f4 lo = {acc[j][ht][0] * rl, acc[j][ht][1] * rl,
               acc[j][ht][2] * rl, acc[j][ht][3] * rl};
      f4 hi = {acc[j][ht][4] * rl, acc[j][ht][5] * rl,
               acc[j][ht][6] * rl, acc[j][ht][7] * rl};
      *(f4*)(out + idx) = lo;
      *(f4*)(out + idx + 4) = hi;
    }
  }
}

// ---------------- launch ----------------

extern "C" void kernel_launch(void* const* d_in, const int* in_sizes, int n_in,
                              void* d_out, int out_size, void* d_ws, size_t ws_size,
                              hipStream_t stream) {
  const float* x = (const float*)d_in[0];
  const unsigned char* pm = (const unsigned char*)d_in[1];
  const float* Wk = (const float*)d_in[2];
  const float* Wq = (const float*)d_in[3];
  const float* Wv = (const float*)d_in[4];
  float* out = (float*)d_out;

  char* ws = (char*)d_ws;
  size_t off = 0;
  unsigned short* xb = (unsigned short*)(ws + off); off += (size_t)B_ * T_ * C_ * 2;
  unsigned short* Qb = (unsigned short*)(ws + off); off += (size_t)B_ * T_ * H_ * 2;
  unsigned short* Kb = (unsigned short*)(ws + off); off += (size_t)B_ * T_ * H_ * 2;
  unsigned short* Vt = (unsigned short*)(ws + off); off += (size_t)B_ * H_ * T_ * 2;
  unsigned short* Wqt = (unsigned short*)(ws + off); off += (size_t)H_ * C_ * 2;
  unsigned short* Wkt = (unsigned short*)(ws + off); off += (size_t)H_ * C_ * 2;
  unsigned short* Wvt = (unsigned short*)(ws + off); off += (size_t)H_ * C_ * 2;

  int nx = B_ * T_ * C_;
  convert_x_kernel<<<nx / 256, 256, 0, stream>>>(x, xb, nx);
  prep_w_kernel<<<(H_ * C_) / 256, 256, 0, stream>>>(Wq, Wk, Wv, Wqt, Wkt, Wvt);
  proj_kernel<<<(B_ * T_ / 16) / 4, 128, 0, stream>>>(xb, Wqt, Wkt, Wvt, Qb, Kb, Vt);
  attn_kernel<<<dim3(T_ / (32 * WAVES_), B_), 128, 0, stream>>>(Qb, Kb, Vt, pm, out);
}
